// SelfAttention2D_21861383536955
// MI455X (gfx1250) — matrix operand order repaired and checked by reading, not compiled, not run on hardware
//
#include <hip/hip_runtime.h>
#include <stdint.h>

// ---------------------------------------------------------------------------
// 2D self-attention with relative position logits (attention-augmented conv)
// B=8, H=W=32, d=256, NH=8, DKH=DVH=32, S=H*W=1024.
//
// All matrix math on v_wmma_f32_16x16x32_bf16 (CDNA5 WMMA, wave32).
// Flash-attention streaming avoids the 256MB logits tensor; rel-position
// logits reduce to per-key-column / per-key-row bias tables RW/RH.
// K/V tiles are staged into LDS with gfx1250 async-to-LDS copies, double
// buffered so the copy of tile kt+1 overlaps compute of tile kt.
// ---------------------------------------------------------------------------

typedef __attribute__((ext_vector_type(16))) __bf16 v16bf;
typedef __attribute__((ext_vector_type(8)))  float  v8f;

#define S_Q   1024
#define NHEAD 8
#define QSCALE 0.17677669529663687f   // 32^-0.5

union ATile { v16bf v; __bf16 h[16]; uint4 q[2]; };

__device__ __forceinline__ v8f wmma_bf16(const ATile& a, const ATile& b, v8f c) {
  return __builtin_amdgcn_wmma_f32_16x16x32_bf16(false, a.v, false, b.v,
                                                 (short)0, c, false, false);
}

// gfx1250 async global->LDS copy (ASYNCcnt-tracked), 16 bytes per lane.
__device__ __forceinline__ void async_ld16(uint32_t lds_addr, const void* gaddr) {
  asm volatile("global_load_async_to_lds_b128 %0, %1, off"
               :: "v"(lds_addr), "v"(gaddr) : "memory");
}
__device__ __forceinline__ void wait_async0() {
  asm volatile("s_wait_asynccnt 0x0" ::: "memory");
}

// --------------------------------------------------------------------------
// Kernel 1: transpose + convert weights to bf16 (Wkqvt: 768x256, Wpt: 256x256)
// --------------------------------------------------------------------------
__global__ void prep_weights_kernel(const float* __restrict__ w_kqv,
                                    const float* __restrict__ w_proj,
                                    __bf16* __restrict__ Wkqvt,
                                    __bf16* __restrict__ Wpt) {
  int i = blockIdx.x * blockDim.x + threadIdx.x;   // 768*256 + 256*256 threads
  if (i < 768 * 256) {
    int f = i >> 8, c = i & 255;
    Wkqvt[i] = (__bf16)w_kqv[c * 768 + f];
  } else {
    int j = i - 768 * 256;
    int f = j >> 8, c = j & 255;
    Wpt[j] = (__bf16)w_proj[c * 256 + f];
  }
}

// --------------------------------------------------------------------------
// Kernel 2: kqv = x @ w_kqv + b ; route to Qh (scaled, bf16 row-major),
// Kh (bf16 row-major), Vt (bf16 TRANSPOSED: [bn, dv, s]).
// One wave = one 16x16 output tile, K-loop of 8 x (K=32) WMMAs.
// --------------------------------------------------------------------------
__global__ void __launch_bounds__(128) kqv_gemm_kernel(
    const float* __restrict__ x, const __bf16* __restrict__ Wt,
    const float* __restrict__ bias,
    __bf16* __restrict__ Qh, __bf16* __restrict__ Kh, __bf16* __restrict__ Vt) {
  const int lane = threadIdx.x & 31;
  const int wv   = threadIdx.x >> 5;
  const int col  = lane & 15;
  const int hi   = lane >> 4;
  const int tile = blockIdx.x * 4 + wv;      // 512 * 48 tiles
  const int ntile = tile % 48;
  const int mtile = tile / 48;
  const int row0  = mtile * 16;
  const int f     = ntile * 16 + col;        // output column of this lane
  const int arow  = row0 + col;              // A-matrix row of this lane

  const float*  xr = x  + (size_t)arow * 256;
  const __bf16* wr = Wt + (size_t)f * 256;

  v8f acc = {0.f,0.f,0.f,0.f,0.f,0.f,0.f,0.f};
#pragma unroll
  for (int kk = 0; kk < 256; kk += 32) {
    ATile a, b;
    // A: fp32 x row -> bf16. elems 0..15 = k in [kk+16*hi, +16) contiguous.
    const float4* p0 = (const float4*)(xr + kk + 16 * hi);
    const float4* p1 = (const float4*)(xr + kk + 16 * hi + 8);
    float4 c0 = p0[0], c1 = p0[1], c2 = p1[0], c3 = p1[1];
    a.h[0]=(__bf16)c0.x; a.h[1]=(__bf16)c0.y; a.h[2]=(__bf16)c0.z; a.h[3]=(__bf16)c0.w;
    a.h[4]=(__bf16)c1.x; a.h[5]=(__bf16)c1.y; a.h[6]=(__bf16)c1.z; a.h[7]=(__bf16)c1.w;
    a.h[8]=(__bf16)c2.x; a.h[9]=(__bf16)c2.y; a.h[10]=(__bf16)c2.z; a.h[11]=(__bf16)c2.w;
    a.h[12]=(__bf16)c3.x; a.h[13]=(__bf16)c3.y; a.h[14]=(__bf16)c3.z; a.h[15]=(__bf16)c3.w;
    // B: Wt row f, 16 contiguous bf16 at k = kk + 16*hi
    const uint4* wp = (const uint4*)(wr + kk + 16 * hi);
    b.q[0] = wp[0];
    b.q[1] = wp[1];
    acc = wmma_bf16(a, b, acc);
  }

  const float bb  = bias[f];
  const int sel   = f >> 8;      // 0 = K, 1 = Q, 2 = V
  const int dd    = f & 255;
  const int head  = dd >> 5;
  const int dch   = dd & 31;
#pragma unroll
  for (int r = 0; r < 8; ++r) {
    const int row = row0 + r + 8 * hi;       // C layout: M = r + 8*hi
    const int bix = row >> 10;
    const int sp  = row & 1023;
    const int bn  = bix * NHEAD + head;
    const float v = acc[r] + bb;
    if (sel == 0) {
      Kh[((size_t)bn * S_Q + sp) * 32 + dch] = (__bf16)v;
    } else if (sel == 1) {
      Qh[((size_t)bn * S_Q + sp) * 32 + dch] = (__bf16)(v * QSCALE);
    } else {
      Vt[((size_t)bn * 32 + dch) * S_Q + sp] = (__bf16)v;
    }
  }
}

// --------------------------------------------------------------------------
// Kernel 3: relative-position bias tables.
//  RW[bn, q=(qr,qc), kc] = Qh[bn, q]        . rel_w[kc - qc + 31]
//  RH[bn, q=(qr,qc), kr] = Qh[bn, qc*32+qr] . rel_h[kr - qr + 31]
// (the transposed q read replicates the reference's missing-transpose.)
// --------------------------------------------------------------------------
__global__ void relbias_kernel(const __bf16* __restrict__ Qh,
                               const float* __restrict__ rel_w,
                               const float* __restrict__ rel_h,
                               float* __restrict__ RW, float* __restrict__ RH) {
  const int idx = blockIdx.x * blockDim.x + threadIdx.x;  // 64*1024*32
  const int kc  = idx & 31;
  const int s   = (idx >> 5) & (S_Q - 1);
  const int bn  = idx >> 15;
  const int qr  = s >> 5, qc = s & 31;

  const __bf16* q1 = Qh + ((size_t)bn * S_Q + s) * 32;
  const float*  w  = rel_w + (kc - qc + 31) * 32;
  float a0 = 0.f;
#pragma unroll
  for (int d = 0; d < 32; ++d) a0 += (float)q1[d] * w[d];
  RW[idx] = a0;

  const __bf16* q2 = Qh + ((size_t)bn * S_Q + (qc * 32 + qr)) * 32;
  const float*  hh = rel_h + (kc - qr + 31) * 32;
  float a1 = 0.f;
#pragma unroll
  for (int d = 0; d < 32; ++d) a1 += (float)q2[d] * hh[d];
  RH[idx] = a1;
}

// --------------------------------------------------------------------------
// Kernel 4: streaming attention. 4 waves/block share one bn; each wave owns
// a 16-query tile. K/V tiles staged in LDS via async copies (double buffer):
//   LDS buf: [0..1023]  = K tile   (32 keys x 32 dkh, row-major, 64B rows)
//            [1024..2047]= V^T tile (32 dv  x 32 keys, row-major, 64B rows)
// S^T = K_tile . Q^T  (query lives in the LANE index -> per-lane softmax),
// O^T = V^T . P^T     (accumulated over 32 key-tile iterations, K=32 each).
// --------------------------------------------------------------------------
__device__ __forceinline__ const char* kv_src(const __bf16* Kb, const __bf16* Vb,
                                              int kk, uint32_t bo) {
  if (bo < 2048u)                         // K region: contiguous 2KB
    return (const char*)Kb + (size_t)kk * 64 + bo;
  const uint32_t vb = bo - 2048u;         // V region: 32 rows, stride 2048B
  return (const char*)Vb + (size_t)(vb >> 6) * 2048 + (size_t)kk * 2 + (vb & 63u);
}

__global__ void __launch_bounds__(128) attn_kernel(
    const __bf16* __restrict__ Qh, const __bf16* __restrict__ Kh,
    const __bf16* __restrict__ Vt, const float* __restrict__ RW,
    const float* __restrict__ RH, __bf16* __restrict__ attn) {
  __shared__ __align__(16) __bf16 sKV[2][2048];   // 8 KB, double buffered

  const int tid  = threadIdx.x;
  const int lane = tid & 31;
  const int wv   = tid >> 5;
  const int col  = lane & 15;
  const int hi   = lane >> 4;
  const int bn    = blockIdx.x >> 4;
  const int qtile = (blockIdx.x & 15) * 4 + wv;
  const int q     = qtile * 16 + col;

  const __bf16* Qb  = Qh + (size_t)bn * S_Q * 32;
  const __bf16* Kb  = Kh + (size_t)bn * S_Q * 32;
  const __bf16* Vb  = Vt + (size_t)bn * 32 * S_Q;
  const float*  RWb = RW + ((size_t)bn * S_Q + q) * 32;
  const float*  RHb = RH + ((size_t)bn * S_Q + q) * 32;
  __builtin_prefetch(RWb, 0, 3);                 // gfx1250 global_prefetch
  __builtin_prefetch(RHb, 0, 3);

  // Async-copy partition: thread tid copies 2x16B of the 4KB K|V region.
  // Waves 0-1 -> K region, waves 2-3 -> V region (uniform per wave).
  const uint32_t bo0 = (uint32_t)tid * 32;
  const uint32_t bo1 = bo0 + 16;
  // Generic LDS pointers are {SHARED_BASE, offset[31:0]} -> low 32 bits = LDS
  // byte offset, which is what the async instruction's VDST wants.
  const uint32_t lds0 = (uint32_t)(uintptr_t)(&sKV[0][0]);
  const uint32_t lds1 = (uint32_t)(uintptr_t)(&sKV[1][0]);

  // B = Q^T (32 x 16): lane holds query q, elems 0..7 = dkh 8*hi.., 8..15 = dkh 16+8*hi...
  ATile bq;
  {
    const __bf16* qp = Qb + (size_t)q * 32;
    bq.q[0] = *(const uint4*)(qp + 8 * hi); bq.q[1] = *(const uint4*)(qp + 16 + 8 * hi);
  }
  // RW row preload: key slots r+8*hi (tile lo16) and 16+r+8*hi (tile hi16).
  float rw0[8], rw1[8];
  {
    const float4* p = (const float4*)(RWb + 8 * hi);
    float4 t0 = p[0], t1 = p[1];
    rw0[0]=t0.x; rw0[1]=t0.y; rw0[2]=t0.z; rw0[3]=t0.w;
    rw0[4]=t1.x; rw0[5]=t1.y; rw0[6]=t1.z; rw0[7]=t1.w;
    const float4* p2 = (const float4*)(RWb + 16 + 8 * hi);
    float4 t2 = p2[0], t3 = p2[1];
    rw1[0]=t2.x; rw1[1]=t2.y; rw1[2]=t2.z; rw1[3]=t2.w;
    rw1[4]=t3.x; rw1[5]=t3.y; rw1[6]=t3.z; rw1[7]=t3.w;
  }

  v8f o0 = {0.f,0.f,0.f,0.f,0.f,0.f,0.f,0.f};   // O^T rows dv 0..15
  v8f o1 = {0.f,0.f,0.f,0.f,0.f,0.f,0.f,0.f};   // O^T rows dv 16..31
  float m = -3.0e38f;
  float lsum = 0.f;

  // Prime buffer 0 with key tile 0.
  async_ld16(lds0 + bo0, kv_src(Kb, Vb, 0, bo0));
  async_ld16(lds0 + bo1, kv_src(Kb, Vb, 0, bo1));

  for (int kt = 0; kt < 32; ++kt) {
    wait_async0();         // my slice of buf[kt&1] has landed
    __syncthreads();       // everyone's slice landed; prev buffer free
    if (kt + 1 < 32) {     // overlap: stage next tile into the other buffer
      const uint32_t lb = (kt & 1) ? lds0 : lds1;
      const int kn = (kt + 1) * 32;
      async_ld16(lb + bo0, kv_src(Kb, Vb, kn, bo0));
      async_ld16(lb + bo1, kv_src(Kb, Vb, kn, bo1));
    }
    const __bf16* sk = &sKV[kt & 1][0];
    const __bf16* sv = &sKV[kt & 1][1024];

    // A = K tiles from LDS (ds_load_b128 x2 per tile).
    ATile ak0, ak1;
    ak0.q[0] = *(const uint4*)(sk + col * 32 + 8 * hi);
    ak0.q[1] = *(const uint4*)(sk + col * 32 + 16 + 8 * hi);
    ak1.q[0] = *(const uint4*)(sk + (16 + col) * 32 + 8 * hi);
    ak1.q[1] = *(const uint4*)(sk + (16 + col) * 32 + 16 + 8 * hi);

    v8f z = {0.f,0.f,0.f,0.f,0.f,0.f,0.f,0.f};
    v8f s0 = wmma_bf16(ak0, bq, z);   // S^T[key = r+8hi,      query = col]
    v8f s1 = wmma_bf16(ak1, bq, z);   // S^T[key = 16+r+8hi,   query = col]

    // add rel biases: kr = kt (constant this iter), kc = key % 32.
    const float rh = RHb[kt];
    float mt = -3.0e38f;
#pragma unroll
    for (int r = 0; r < 8; ++r) {
      s0[r] += rw0[r] + rh;
      s1[r] += rw1[r] + rh;
      mt = fmaxf(mt, fmaxf(s0[r], s1[r]));
    }
    // online softmax: stats per query = per lane (+ partner half via xor16)
    mt = fmaxf(mt, __shfl_xor(mt, 16, 32));
    const float mnew  = fmaxf(m, mt);
    const float alpha = __expf(m - mnew);
    float p0[8], p1[8];
    float ssum = 0.f;
#pragma unroll
    for (int r = 0; r < 8; ++r) {
      p0[r] = __expf(s0[r] - mnew);
      p1[r] = __expf(s1[r] - mnew);
      ssum += p0[r] + p1[r];
    }
    ssum += __shfl_xor(ssum, 16, 32);
    lsum = lsum * alpha + ssum;
    m = mnew;

    // Build B = P^T (32 keys x 16 queries): lane needs keys 8*hi+r and
    // 16+8*hi+r, both held by its own half.
    ATile bp;
#pragma unroll
    for (int r = 0; r < 8; ++r) {
      const float x0 = __shfl_xor(p0[r], 16, 32);
      const float x1 = __shfl_xor(p1[r], 16, 32);
      bp.h[r]     = (__bf16)p0[r];   // keys 8*hi + r
      bp.h[8 + r] = (__bf16)p1[r];   // keys 16 + 8*hi + r
    }

    // A = V^T tiles from LDS.
    ATile av0, av1;
    av0.q[0] = *(const uint4*)(sv + col * 32 + 8 * hi);
    av0.q[1] = *(const uint4*)(sv + col * 32 + 16 + 8 * hi);
    av1.q[0] = *(const uint4*)(sv + (16 + col) * 32 + 8 * hi);
    av1.q[1] = *(const uint4*)(sv + (16 + col) * 32 + 16 + 8 * hi);

#pragma unroll
    for (int r = 0; r < 8; ++r) { o0[r] *= alpha; o1[r] *= alpha; }
    o0 = wmma_bf16(av0, bp, o0);
    o1 = wmma_bf16(av1, bp, o1);
  }

  const float inv = 1.0f / lsum;
  union { uint4 u; __bf16 h[8]; } st0, st1;
#pragma unroll
  for (int r = 0; r < 8; ++r) {
    st0.h[r] = (__bf16)(o0[r] * inv);   // dv = r + 8*hi
    st1.h[r] = (__bf16)(o1[r] * inv);   // dv = 16 + r + 8*hi
  }
  const int bix = bn >> 3, head = bn & 7;
  __bf16* op = attn + (((size_t)bix * S_Q + q) * 256 + head * 32);
  *(uint4*)(op + 8 * hi)      = st0.u;
  *(uint4*)(op + 16 + 8 * hi) = st1.u;
}

// --------------------------------------------------------------------------
// Kernel 5: out = attn @ w_proj + b_proj (8192x256x256), fp32 output.
// --------------------------------------------------------------------------
__global__ void __launch_bounds__(128) proj_gemm_kernel(
    const __bf16* __restrict__ attn, const __bf16* __restrict__ Wpt,
    const float* __restrict__ bias, float* __restrict__ out) {
  const int lane = threadIdx.x & 31;
  const int wv   = threadIdx.x >> 5;
  const int col  = lane & 15;
  const int hi   = lane >> 4;
  const int tile  = blockIdx.x * 4 + wv;   // 512 * 16 tiles
  const int ntile = tile & 15;
  const int mtile = tile >> 4;
  const int row0  = mtile * 16;
  const int f     = ntile * 16 + col;

  const __bf16* ar = attn + (size_t)(row0 + col) * 256;
  const __bf16* wr = Wpt + (size_t)f * 256;

  v8f acc = {0.f,0.f,0.f,0.f,0.f,0.f,0.f,0.f};
#pragma unroll
  for (int kk = 0; kk < 256; kk += 32) {
    ATile a, b;
    a.q[0] = *(const uint4*)(ar + kk + 16 * hi);
    a.q[1] = *(const uint4*)(ar + kk + 16 * hi + 8);
    const uint4* wp = (const uint4*)(wr + kk + 16 * hi);
    b.q[0] = wp[0];
    b.q[1] = wp[1];
    acc = wmma_bf16(a, b, acc);
  }
  const float bb = bias[f];
#pragma unroll
  for (int r = 0; r < 8; ++r) {
    const int row = row0 + r + 8 * hi;
    out[(size_t)row * 256 + f] = acc[r] + bb;
  }
}

// --------------------------------------------------------------------------
// Launch. Workspace layout (~33 MB): Qh|Kh|Vt|attn (bf16), RW|RH (f32),
// Wkqvt|Wpt (bf16).
// --------------------------------------------------------------------------
extern "C" void kernel_launch(void* const* d_in, const int* in_sizes, int n_in,
                              void* d_out, int out_size, void* d_ws, size_t ws_size,
                              hipStream_t stream) {
  (void)in_sizes; (void)n_in; (void)out_size; (void)ws_size;
  const float* x      = (const float*)d_in[0];
  const float* w_kqv  = (const float*)d_in[1];
  const float* b_kqv  = (const float*)d_in[2];
  const float* w_proj = (const float*)d_in[3];
  const float* b_proj = (const float*)d_in[4];
  const float* rel_w  = (const float*)d_in[5];
  const float* rel_h  = (const float*)d_in[6];
  float* out = (float*)d_out;

  char* ws = (char*)d_ws;
  size_t off = 0;
  auto take = [&](size_t bytes) -> char* {
    char* p = ws + off;
    off += (bytes + 255) & ~(size_t)255;
    return p;
  };
  __bf16* Qh    = (__bf16*)take((size_t)64 * S_Q * 32 * 2);   // 4 MB
  __bf16* Kh    = (__bf16*)take((size_t)64 * S_Q * 32 * 2);   // 4 MB
  __bf16* Vt    = (__bf16*)take((size_t)64 * 32 * S_Q * 2);   // 4 MB
  __bf16* attnb = (__bf16*)take((size_t)8192 * 256 * 2);      // 4 MB
  float*  RW    = (float*)take((size_t)64 * S_Q * 32 * 4);    // 8 MB
  float*  RH    = (float*)take((size_t)64 * S_Q * 32 * 4);    // 8 MB
  __bf16* Wkqvt = (__bf16*)take((size_t)768 * 256 * 2);
  __bf16* Wpt   = (__bf16*)take((size_t)256 * 256 * 2);

  prep_weights_kernel<<<1024, 256, 0, stream>>>(w_kqv, w_proj, Wkqvt, Wpt);
  kqv_gemm_kernel<<<6144, 128, 0, stream>>>(x, Wkqvt, b_kqv, Qh, Kh, Vt);
  relbias_kernel<<<8192, 256, 0, stream>>>(Qh, rel_w, rel_h, RW, RH);
  attn_kernel<<<1024, 128, 0, stream>>>(Qh, Kh, Vt, RW, RH, attnb);
  proj_gemm_kernel<<<2048, 128, 0, stream>>>(attnb, Wpt, b_proj, out);
}
